// DepthwiseSeparableConv_50431505990045
// MI455X (gfx1250) — 8-bit matrix operand order repaired and checked by reading, compiled, not run on hardware
//
#include <hip/hip_runtime.h>
#include <hip/hip_bf16.h>
#include <cstddef>


typedef int v8i __attribute__((ext_vector_type(8)));

constexpr int  B_    = 16;
constexpr int  CIN   = 256;
constexpr int  COUT  = 512;
constexpr int  H_    = 112;
constexpr int  W_    = 112;
constexpr int  HW    = H_ * W_;
constexpr long NTOT  = (long)B_ * HW;
constexpr long XTOT  = (long)B_ * CIN * HW;
constexpr long YTOT  = (long)B_ * COUT * HW;

constexpr size_t WS_WC = 256;
constexpr size_t WS_HQ = 256 + (size_t)COUT * CIN;


__device__ __forceinline__ float pow2scale(float t, float qmax) {
  return exp2f(ceilf(log2f(t / qmax + 1e-12f)));
}

__device__ __forceinline__ void block_absmax_atomic_f(float m, unsigned* slot) {
  __shared__ float s_red[8];
  #pragma unroll
  for (int o = 16; o > 0; o >>= 1) m = fmaxf(m, __shfl_xor(m, o, 32));
  int lane = threadIdx.x & 31, wid = threadIdx.x >> 5;
  if (lane == 0) s_red[wid] = m;
  __syncthreads();
  if (threadIdx.x < 32) {
    float v = (threadIdx.x < 8) ? s_red[threadIdx.x] : 0.f;
    #pragma unroll
    for (int o = 4; o > 0; o >>= 1) v = fmaxf(v, __shfl_xor(v, o, 32));
    if (threadIdx.x == 0) atomicMax(slot, __float_as_uint(v));
  }
}

__global__ void k_init(unsigned* us) {
  if (threadIdx.x < 16) us[threadIdx.x] = 0u;
}

__global__ __launch_bounds__(256) void k_absmax(const float* __restrict__ src,
                                                long n, unsigned* slot) {
  float m = 0.f;
  long stride = (long)gridDim.x * blockDim.x;
  for (long i = (long)blockIdx.x * blockDim.x + threadIdx.x; i < n; i += stride)
    m = fmaxf(m, fabsf(src[i]));
  block_absmax_atomic_f(m, slot);
}

__global__ void k_scales1(unsigned* us) {
  if (threadIdx.x == 0) {
    float* fs = (float*)us;
    fs[6] = pow2scale(__uint_as_float(us[0]), 7.f);
    fs[7] = pow2scale(__uint_as_float(us[1]), 7.f);
    fs[8] = pow2scale(__uint_as_float(us[2]), 7.f);
  }
}

__device__ __forceinline__ int dw_acc_int(const float* __restrict__ x,
                                          const float* __restrict__ wdw,
                                          float inv_sx, float inv_sw,
                                          int b, int c, int yy, int xx) {
  const float* xp = x + ((size_t)(b * CIN + c)) * HW;
  const float* wp = wdw + c * 9;
  int acc = 0;
  #pragma unroll
  for (int dy = 0; dy < 3; ++dy) {
    int y2 = yy + dy - 1;
    if ((unsigned)y2 >= (unsigned)H_) continue;
    #pragma unroll
    for (int dx = 0; dx < 3; ++dx) {
      int x2 = xx + dx - 1;
      if ((unsigned)x2 >= (unsigned)W_) continue;
      int cx = min(7, max(-8, __float2int_rn(xp[y2 * W_ + x2] * inv_sx)));
      int cw = min(7, max(-7, __float2int_rn(wp[dy * 3 + dx] * inv_sw)));
      acc += cx * cw;
    }
  }
  return acc;
}

__global__ __launch_bounds__(256) void k_dw_absmax(const float* __restrict__ x,
                                                   const float* __restrict__ wdw,
                                                   unsigned* us) {
  const float* fs = (const float*)us;
  float sx = fs[6], sw = fs[7];
  float inv_sx = 1.f / sx, inv_sw = 1.f / sw, sxw = sx * sw;
  float m = 0.f;
  long stride = (long)gridDim.x * blockDim.x;
  for (long idx = (long)blockIdx.x * blockDim.x + threadIdx.x; idx < XTOT; idx += stride) {
    int hw = (int)(idx % HW);
    int t  = (int)(idx / HW);
    int c = t & (CIN - 1), b = t >> 8;
    float raw = (float)dw_acc_int(x, wdw, inv_sx, inv_sw, b, c, hw / W_, hw % W_) * sxw;
    m = fmaxf(m, fabsf(raw));
  }
  block_absmax_atomic_f(m, us + 3);
}

__global__ void k_scale_h(unsigned* us) {
  if (threadIdx.x == 0) {
    float* fs = (float*)us;
    fs[9] = pow2scale(__uint_as_float(us[3]), 7.f);
  }
}

__global__ __launch_bounds__(256) void k_dw_quant(const float* __restrict__ x,
                                                  const float* __restrict__ wdw,
                                                  signed char* __restrict__ hq,
                                                  unsigned* us) {
  __shared__ signed char sm[256 * 33];
  const float* fs = (const float*)us;
  float sx = fs[6], sw = fs[7], sh = fs[9];
  float inv_sx = 1.f / sx, inv_sw = 1.f / sw;
  float raw2code = (sx * sw) / sh;
  int b   = blockIdx.x / (HW / 32);
  int hw0 = (blockIdx.x % (HW / 32)) * 32;
  int lane = threadIdx.x & 31, wid = threadIdx.x >> 5;
  float mc = 0.f;
  #pragma unroll 4
  for (int it = 0; it < 32; ++it) {
    int c  = it * 8 + wid;
    int hw = hw0 + lane;
    int acc = dw_acc_int(x, wdw, inv_sx, inv_sw, b, c, hw / W_, hw % W_);
    int q = min(7, max(-8, __float2int_rn((float)acc * raw2code)));
    sm[c * 33 + lane] = (signed char)q;
    mc = fmaxf(mc, (float)abs(q));
  }
  block_absmax_atomic_f(mc, us + 4);
  for (int it = 0; it < 32; ++it) {
    hq[((size_t)(b * HW + hw0 + it)) * CIN + threadIdx.x] = sm[threadIdx.x * 33 + it];
  }
}

__global__ void k_scales2(unsigned* us) {
  if (threadIdx.x == 0) {
    float* fs = (float*)us;
    float t2 = fs[9] * __uint_as_float(us[4]);
    fs[10] = pow2scale(t2, 7.f);
    fs[11] = fs[9] / fs[10];
    fs[13] = fs[10] * fs[8];
  }
}

__global__ __launch_bounds__(256) void k_requant(unsigned* __restrict__ hq4,
                                                 const unsigned* us, long n4) {
  const float* fs = (const float*)us;
  float ratio = fs[11];
  long stride = (long)gridDim.x * blockDim.x;
  for (long i = (long)blockIdx.x * blockDim.x + threadIdx.x; i < n4; i += stride) {
    unsigned w = hq4[i], r = 0;
    #pragma unroll
    for (int k = 0; k < 4; ++k) {
      int c = (int)(signed char)(w >> (8 * k));
      int q = min(7, max(-8, __float2int_rn((float)c * ratio)));
      r |= ((unsigned)(q & 0xff)) << (8 * k);
    }
    hq4[i] = r;
  }
}

__global__ __launch_bounds__(256) void k_quant_wpw(const float* __restrict__ wpw,
                                                   signed char* __restrict__ wc,
                                                   const unsigned* us) {
  const float* fs = (const float*)us;
  float inv = 1.f / fs[8];
  int i = blockIdx.x * 256 + threadIdx.x;
  wc[i] = (signed char)min(7, max(-7, __float2int_rn(wpw[i] * inv)));
}

__global__ __launch_bounds__(256) void k_gemm(const signed char* __restrict__ hq,
                                              const signed char* __restrict__ wc,
                                              unsigned* us,
                                              float* __restrict__ out) {
  __shared__ int ldsA[8192];
  const float* fs = (const float*)us;
  const int m0 = blockIdx.y * 128;
  const int n0 = blockIdx.x * 128;
  const int tid = threadIdx.x, lane = tid & 31, wid = tid >> 5;
  const int wm = wid >> 2, wn = wid & 3;

  const int* w32 = (const int*)wc;
  #pragma unroll 4
  for (int it = 0; it < 32; ++it) {
    int d   = it * 256 + tid;
    int tm  = d >> 10;
    int kc  = (d >> 8) & 3;
    int lf  = (d >> 3) & 31;
    int v   = d & 7;
    int M   = m0 + tm * 16 + (lf & 15);
    int K   = kc * 64 + ((v & 4) ? 32 : 0) + (v & 3) * 4
              + ((lf & 16) ? 16 : 0);
    ldsA[d] = w32[M * (CIN / 4) + (K >> 2)];
  }
  __syncthreads();

  v8i acc[4][2] = {};

  #pragma unroll
  for (int kc = 0; kc < 4; ++kc) {
    v8i bfr[2];
    #pragma unroll
    for (int nf = 0; nf < 2; ++nf) {
      size_t nn   = (size_t)(n0 + wn * 32 + nf * 16 + (lane & 15));
      size_t base = nn * CIN + (size_t)kc * 64 + ((lane & 16) ? 16 : 0);
      int4 lo = *reinterpret_cast<const int4*>(hq + base);
      int4 hi = *reinterpret_cast<const int4*>(hq + base + 32);
      bfr[nf] = (v8i){lo.x, lo.y, lo.z, lo.w, hi.x, hi.y, hi.z, hi.w};
      if (kc < 3) __builtin_prefetch(hq + base + 64, 0, 1);
    }
    v8i afr[4];
    #pragma unroll
    for (int mi = 0; mi < 4; ++mi) {
      const int4* p = reinterpret_cast<const int4*>(
          &ldsA[((wm * 4 + mi) * 4 + kc) * 256 + lane * 8]);
      int4 a0 = p[0], a1 = p[1];
      afr[mi] = (v8i){a0.x, a0.y, a0.z, a0.w, a1.x, a1.y, a1.z, a1.w};
    }
    #pragma unroll
    for (int mi = 0; mi < 4; ++mi)
      #pragma unroll
      for (int nf = 0; nf < 2; ++nf)
        acc[mi][nf] = __builtin_amdgcn_wmma_i32_16x16x64_iu8(
            true, afr[mi], true, bfr[nf], acc[mi][nf], false, false);
  }

  const float osc = fs[13];
  float lmax = 0.f;
  const int nlo = lane & 15;
  const int mhi = (lane >> 4) * 8;
  #pragma unroll
  for (int mi = 0; mi < 4; ++mi) {
    int mbase = m0 + (wm * 4 + mi) * 16 + mhi;
    #pragma unroll
    for (int nf = 0; nf < 2; ++nf) {
      int n    = n0 + wn * 32 + nf * 16 + nlo;
      int bimg = n / HW;
      int hw   = n - bimg * HW;
      size_t obase = ((size_t)bimg * COUT) * HW + hw;
      #pragma unroll
      for (int r = 0; r < 8; ++r) {
        float y = (float)acc[mi][nf][r] * osc;
        out[obase + (size_t)(mbase + r) * HW] = y;
        lmax = fmaxf(lmax, fabsf(y));
      }
    }
  }
  block_absmax_atomic_f(lmax, us + 5);
}

__global__ void k_scale_y(unsigned* us) {
  if (threadIdx.x == 0) {
    float* fs = (float*)us;
    fs[12] = pow2scale(__uint_as_float(us[5]), 7.f);
  }
}

__global__ __launch_bounds__(256) void k_final(float* __restrict__ out,
                                               const unsigned* us, long n) {
  const float* fs = (const float*)us;
  float s = fs[12], inv = 1.f / s;
  long stride = (long)gridDim.x * blockDim.x;
  for (long i = (long)blockIdx.x * blockDim.x + threadIdx.x; i < n; i += stride) {
    int q = min(7, max(-8, __float2int_rn(out[i] * inv)));
    out[i] = (float)q * s;
  }
}

extern "C" void kernel_launch(void* const* d_in, const int* in_sizes, int n_in,
                              void* d_out, int out_size, void* d_ws, size_t ws_size,
                              hipStream_t stream) {
  const float* x    = (const float*)d_in[0];
  const float* w_dw = (const float*)d_in[1];
  const float* w_pw = (const float*)d_in[2];
  float* out        = (float*)d_out;

  unsigned*    us = (unsigned*)d_ws;
  signed char* wc = (signed char*)d_ws + WS_WC;
  signed char* hq = (signed char*)d_ws + WS_HQ;

  k_init<<<1, 32, 0, stream>>>(us);

  k_absmax<<<4096, 256, 0, stream>>>(x,    XTOT,                 us + 0);
  k_absmax<<<9,    256, 0, stream>>>(w_dw, (long)CIN * 9,        us + 1);
  k_absmax<<<512,  256, 0, stream>>>(w_pw, (long)COUT * CIN,     us + 2);
  k_scales1<<<1, 1, 0, stream>>>(us);

  k_dw_absmax<<<25088, 256, 0, stream>>>(x, w_dw, us);
  k_scale_h<<<1, 1, 0, stream>>>(us);

  k_dw_quant<<<B_ * (HW / 32), 256, 0, stream>>>(x, w_dw, hq, us);
  k_scales2<<<1, 1, 0, stream>>>(us);

  k_requant<<<8192, 256, 0, stream>>>((unsigned*)hq, us, XTOT / 4);
  k_quant_wpw<<<(COUT * CIN) / 256, 256, 0, stream>>>(w_pw, wc, us);

  dim3 ggrid((unsigned)(NTOT / 128), COUT / 128);
  k_gemm<<<ggrid, 256, 0, stream>>>(hq, wc, us, out);

  k_scale_y<<<1, 1, 0, stream>>>(us);
  k_final<<<25088, 256, 0, stream>>>(out, us, YTOT);
}
